// highwayNet_8538394984640
// MI455X (gfx1250) — compile-verified, matrix operand order checked by reading, not run on hardware
//
#include <hip/hip_runtime.h>

typedef __attribute__((ext_vector_type(16))) _Float16 v16h;
typedef __attribute__((ext_vector_type(8)))  float    v8f;

#define LRELU(v) ((v) > 0.f ? (v) : 0.1f * (v))

// K-offset of element e (0..15) of a 16-bit WMMA A/B fragment for this lane,
// per CDNA5 ISA 7.12.2 (16-bit A-matrix 16x32 layout; B mirrors with N<->M).
// For a fixed lane this yields two contiguous 16-byte runs -> b128 loads.
__device__ inline int frag_k(int lane, int e) {
  int p = e >> 1;
  return 2 * p + ((p >= 4) ? 8 : 0) + ((lane >> 4) & 1) * 8 + (e & 1);
}

// Load a 16xK A-tile (row-major halves, row stride ldk) into fragment regs.
template <typename PtrT>
__device__ inline v16h load_a(PtrT src, int ldk, int kt) {
  int lane = threadIdx.x & 31;
  int row = lane & 15;
  v16h a;
#pragma unroll
  for (int e = 0; e < 16; ++e)
    a[e] = src[row * ldk + kt * 32 + frag_k(lane, e)];
  return a;
}

// Load a pre-swizzled B fragment: each lane's 16 halves are contiguous.
__device__ inline v16h load_b(const _Float16* packed, int nt, int kt, int nKt) {
  int lane = threadIdx.x & 31;
  return *(const v16h*)(packed + (size_t)(nt * nKt + kt) * 512 + lane * 16);
}

__device__ inline v8f wmma_f16(v16h a, v16h b, v8f c) {
  return __builtin_amdgcn_wmma_f32_16x16x32_f16(false, a, false, b, (short)0, c,
                                                false, false);
}

// C/D layout: vgpr r, lanes 0-15 -> (M=r, N=lane); lanes 16-31 -> (M=r+8, N=lane-16).
// Add bias, leaky-relu, store as f16 to LDS (row-major, stride ldn halves).
__device__ inline void store_c(_Float16* dst, int ldn, int colBase, v8f c,
                               const float* bias, int nt) {
  int lane = threadIdx.x & 31;
  int n = lane & 15;
  int mo = ((lane >> 4) & 1) * 8;
  float bv = bias[nt * 16 + n];
#pragma unroll
  for (int r = 0; r < 8; ++r) {
    float v = c[r] + bv;
    v = LRELU(v);
    dst[(r + mo) * ldn + colBase + nt * 16 + n] = (_Float16)v;
  }
}

// ---------------------------------------------------------------------------
// Kernel 1 (single launch): swizzle ALL weights into v16h B fragments.
// grid = 46 waves; blockIdx.x selects (weight, tile).
// pw map (halves): Wh1@0(2t) Wh2@1024(4t) Wf1@3072(4t,nKt2) Wf2@5120(4t)
//                  Wdyn@7168(4t,nKt2) Wnbr@9216(4t) Wout@11264(24t,nKt3)
// ---------------------------------------------------------------------------
__global__ __launch_bounds__(32) void hw_pack_all(
    const float* __restrict__ W_h1, const float* __restrict__ W_h2,
    const float* __restrict__ W_f1, const float* __restrict__ W_f2,
    const float* __restrict__ W_dyn, const float* __restrict__ W_nbr,
    const float* __restrict__ W_out, _Float16* __restrict__ pw) {
  int id = blockIdx.x;
  const float* W;
  int Kin, Nout, nKt, base, t0;
  if (id < 2)       { W = W_h1;  Kin = 32; Nout = 32;  nKt = 1; base = 0;     t0 = 0; }
  else if (id < 6)  { W = W_h2;  Kin = 32; Nout = 64;  nKt = 1; base = 1024;  t0 = 2; }
  else if (id < 10) { W = W_f1;  Kin = 50; Nout = 32;  nKt = 2; base = 3072;  t0 = 6; }
  else if (id < 14) { W = W_f2;  Kin = 32; Nout = 64;  nKt = 1; base = 5120;  t0 = 10; }
  else if (id < 18) { W = W_dyn; Kin = 64; Nout = 32;  nKt = 2; base = 7168;  t0 = 14; }
  else if (id < 22) { W = W_nbr; Kin = 32; Nout = 64;  nKt = 1; base = 9216;  t0 = 18; }
  else              { W = W_out; Kin = 96; Nout = 128; nKt = 3; base = 11264; t0 = 22; }
  int tile = id - t0;
  int kt = tile % nKt, nt = tile / nKt;
  int lane = threadIdx.x;
  int n = nt * 16 + (lane & 15);
  _Float16* o = pw + base + (size_t)tile * 512 + lane * 16;
#pragma unroll
  for (int e = 0; e < 16; ++e) {
    int k = kt * 32 + frag_k(lane, e);
    float v = (k < Kin && n < Nout) ? W[k * Nout + n] : 0.f;
    o[e] = (_Float16)v;
  }
}

// ---------------------------------------------------------------------------
// Kernel 2 (single launch): fully fused pipeline. One 256-thread block
// (8 wave32) handles 16 scenes: normalization -> LDS features -> WMMA chain.
// Intermediates never touch HBM. Stage work is split across waves by N-tile;
// all branches are wave-uniform so EXEC is all-ones around every WMMA.
// ---------------------------------------------------------------------------
__global__ __launch_bounds__(256) void hw_fused(
    const float* __restrict__ scene, const _Float16* __restrict__ pw,
    const float* __restrict__ b_h1, const float* __restrict__ b_h2,
    const float* __restrict__ b_f1, const float* __restrict__ b_f2,
    const float* __restrict__ b_dyn, const float* __restrict__ b_nbr,
    const float* __restrict__ b_out, const float* __restrict__ w_op,
    const float* __restrict__ b_op, float* __restrict__ out, int Bn) {
  __shared__ _Float16 relL[15 * 16 * 32];  // [j][row][32]
  __shared__ _Float16 histL[16 * 32];
  __shared__ _Float16 futL[16 * 64];
  __shared__ _Float16 s0[16 * 32];  // hist h1 out
  __shared__ _Float16 s1[16 * 64];  // hist h2 out
  __shared__ _Float16 s2[16 * 32];  // fut f1 out
  __shared__ _Float16 s3[16 * 64];  // pooled nbr
  __shared__ _Float16 s4[16 * 64];  // fut f2 out
  __shared__ _Float16 enc[16 * 96];
  __shared__ float feat[16 * 128];

  const int g = blockIdx.x;
  const int lane = threadIdx.x & 31;
  const int wid = threadIdx.x >> 5;

  const _Float16* Wh1p  = pw;
  const _Float16* Wh2p  = pw + 1024;
  const _Float16* Wf1p  = pw + 3072;
  const _Float16* Wf2p  = pw + 5120;
  const _Float16* Wdynp = pw + 7168;
  const _Float16* Wnbrp = pw + 9216;
  const _Float16* Woutp = pw + 11264;

  const v8f zero = {};

  // ===== Prep: wave w normalizes scenes (rows) 2w and 2w+1 into LDS =====
  for (int rr = 0; rr < 2; ++rr) {
    const int r = wid * 2 + rr;  // row in group, 0..15
    const float* sc = scene + (size_t)(g * 16 + r) * 16 * 82;
    const float refx = sc[15];       // hero (agent 0), ch 0, t = T_H
    const float refy = sc[41 + 15];  // hero, ch 1

    // abs-max over 16 agents x 16 timesteps per channel (wave32 reduction)
    int a = lane & 15;
    int t0 = (lane >> 4) * 8;
    float mx = 0.f, my = 0.f;
    for (int t = t0; t < t0 + 8; ++t) {
      mx = fmaxf(mx, fabsf(sc[a * 82 + t] - refx));
      my = fmaxf(my, fabsf(sc[a * 82 + 41 + t] - refy));
    }
    for (int off = 16; off >= 1; off >>= 1) {
      mx = fmaxf(mx, __shfl_xor(mx, off, 32));
      my = fmaxf(my, __shfl_xor(my, off, 32));
    }

    const int t = lane >> 1, c = lane & 1;
    const float refc = c ? refy : refx;
    const float scc = c ? my : mx;

    histL[r * 32 + lane] = (_Float16)((sc[c * 41 + t] - refc) / scc);

    for (int l = lane; l < 64; l += 32) {  // fut, zero-padded 50 -> 64
      float v = 0.f;
      if (l < 50) {
        int tt = l >> 1, cc = l & 1;
        v = (sc[cc * 41 + 16 + tt] - (cc ? refy : refx)) / (cc ? my : mx);
      }
      futL[r * 64 + l] = (_Float16)v;
    }

    float hv = sc[c * 41 + t];  // rel = (hero - nbr)/scale (ref cancels)
    for (int j = 1; j <= 15; ++j) {
      float nv = sc[j * 82 + c * 41 + t];
      relL[(j - 1) * 512 + r * 32 + lane] = (_Float16)((hv - nv) / scc);
    }
  }
  __syncthreads();

  // ===== Stage 1: hist-h1 (waves 0-1) | fut-f1 (2-3) | nbr pool (4-7) =====
  if (wid < 2) {
    v16h a = load_a(histL, 32, 0);
    store_c(s0, 32, 0, wmma_f16(a, load_b(Wh1p, wid, 0, 1), zero), b_h1, wid);
  } else if (wid < 4) {
    int nt = wid - 2;
    v8f d = wmma_f16(load_a(futL, 64, 0), load_b(Wf1p, nt, 0, 2), zero);
    d = wmma_f16(load_a(futL, 64, 1), load_b(Wf1p, nt, 1, 2), d);
    store_c(s2, 32, 0, d, b_f1, nt);
  } else {
    int nt = wid - 4;
    v16h wb = load_b(Wnbrp, nt, 0, 1);
    v8f m;
#pragma unroll
    for (int r = 0; r < 8; ++r) m[r] = -3.0e38f;
    for (int j = 0; j < 15; ++j) {
      v8f d = wmma_f16(load_a(relL + j * 512, 32, 0), wb, zero);
#pragma unroll
      for (int r = 0; r < 8; ++r) m[r] = fmaxf(m[r], d[r]);
    }
    // lrelu(max + b) == max(lrelu(x + b)): lrelu monotone, bias per-column
    store_c(s3, 64, 0, m, b_nbr, nt);
  }
  __syncthreads();

  // ===== Stage 2: hist-h2 (waves 0-3) | fut-f2 (waves 4-7) =====
  if (wid < 4) {
    store_c(s1, 64, 0, wmma_f16(load_a(s0, 32, 0), load_b(Wh2p, wid, 0, 1), zero),
            b_h2, wid);
  } else {
    int nt = wid - 4;
    store_c(s4, 64, 0, wmma_f16(load_a(s2, 32, 0), load_b(Wf2p, nt, 0, 1), zero),
            b_f2, nt);
  }
  __syncthreads();

  // ===== Stage 3: shared dyn layer on hist/pooled/fut (waves 0-5) =====
  if (wid < 6) {
    const _Float16* src = (wid < 2) ? s1 : (wid < 4) ? s3 : s4;
    int colBase = (wid < 2) ? 0 : (wid < 4) ? 32 : 64;
    int nt = wid & 1;
    v8f d = wmma_f16(load_a(src, 64, 0), load_b(Wdynp, nt, 0, 2), zero);
    d = wmma_f16(load_a(src, 64, 1), load_b(Wdynp, nt, 1, 2), d);
    store_c(enc, 96, colBase, d, b_dyn, nt);
  }
  __syncthreads();

  // ===== Stage 4: feature = lrelu(enc @ W_out + b_out), all 8 waves =====
  {
    float* outFeat = out + Bn;  // d_out = [logit(B)] ++ [feature(Bx128)]
    int nt = wid;
    v8f d = zero;
#pragma unroll
    for (int kt = 0; kt < 3; ++kt)
      d = wmma_f16(load_a(enc, 96, kt), load_b(Woutp, nt, kt, 3), d);
    int n = lane & 15;
    int mo = ((lane >> 4) & 1) * 8;
    float bv = b_out[nt * 16 + n];
#pragma unroll
    for (int r = 0; r < 8; ++r) {
      float v = d[r] + bv;
      v = LRELU(v);
      feat[(r + mo) * 128 + nt * 16 + n] = v;
      outFeat[(size_t)(g * 16 + r + mo) * 128 + nt * 16 + n] = v;
    }
  }
  __syncthreads();

  // ===== Stage 5: logit = sigmoid(feat @ W_op + b_op), 16 lanes/row =====
  {
    int row = threadIdx.x >> 4;  // 16 groups of 16 lanes
    int l4 = threadIdx.x & 15;
    float s = 0.f;
    for (int k = l4; k < 128; k += 16) s += feat[row * 128 + k] * w_op[k];
    for (int off = 8; off >= 1; off >>= 1) s += __shfl_xor(s, off, 16);
    if (l4 == 0) {
      float tot = s + b_op[0];
      out[g * 16 + row] = 1.f / (1.f + expf(-tot));
    }
  }
}

// ---------------------------------------------------------------------------
extern "C" void kernel_launch(void* const* d_in, const int* in_sizes, int n_in,
                              void* d_out, int out_size, void* d_ws,
                              size_t ws_size, hipStream_t stream) {
  const int Bn = 16384;
  const float* scene = (const float*)d_in[0];
  const float* W_h1 = (const float*)d_in[3];
  const float* b_h1 = (const float*)d_in[4];
  const float* W_h2 = (const float*)d_in[5];
  const float* b_h2 = (const float*)d_in[6];
  const float* W_f1 = (const float*)d_in[7];
  const float* b_f1 = (const float*)d_in[8];
  const float* W_f2 = (const float*)d_in[9];
  const float* b_f2 = (const float*)d_in[10];
  const float* W_dyn = (const float*)d_in[11];
  const float* b_dyn = (const float*)d_in[12];
  const float* W_nbr = (const float*)d_in[13];
  const float* b_nbr = (const float*)d_in[14];
  const float* W_out = (const float*)d_in[15];
  const float* b_out = (const float*)d_in[16];
  const float* W_op = (const float*)d_in[17];
  const float* b_op = (const float*)d_in[18];

  _Float16* pw = (_Float16*)d_ws;  // 23552 halves = 47 KB of scratch

  hw_pack_all<<<46, 32, 0, stream>>>(W_h1, W_h2, W_f1, W_f2, W_dyn, W_nbr,
                                     W_out, pw);

  hw_fused<<<Bn / 16, 256, 0, stream>>>(scene, pw, b_h1, b_h2, b_f1, b_f2,
                                        b_dyn, b_nbr, b_out, W_op, b_op,
                                        (float*)d_out, Bn);
}
